// DirModelToFace_64768106824135
// MI455X (gfx1250) — compile-verified, matrix operand order checked by reading, not run on hardware
//
#include <hip/hip_runtime.h>
#include <hip/hip_bf16.h>

// ---------------------------------------------------------------------------
// DirModelToFace on MI455X (gfx1250, wave32).
// Strategy:
//  - sparse quaternion message passing: wave-per-(edge,batch), 32-float
//    segments, global_atomic_add_f32 scatter (memory-bound, dominates).
//  - conv1x1_prebn: fold BN scale into W (bf16), shift into bias, then
//    bf16 WMMA GEMM (v_wmma_f32_16x16x32_bf16), A tile staged in LDS.
// ---------------------------------------------------------------------------

typedef __attribute__((ext_vector_type(16))) __bf16 v16bf;
typedef __attribute__((ext_vector_type(8)))  float  v8f;

struct __align__(16) U4 { unsigned int a, b, c, d; };
union FragBF { v16bf v; U4 q[2]; };

__device__ __forceinline__ float eluf(float x) {
    return x > 0.f ? x : expm1f(x);
}
__device__ __forceinline__ unsigned short f2bf(float f) {
    unsigned int u = __float_as_uint(f);
    u += 0x7FFFu + ((u >> 16) & 1u);   // round to nearest even
    return (unsigned short)(u >> 16);
}

// ---------------------------------------------------------------------------
__global__ void zero_f32(float* __restrict__ p, long n) {
    long i = (long)blockIdx.x * blockDim.x + threadIdx.x;
    if (i < n) p[i] = 0.f;
}

// v = inputs @ W1 + b1   (K=3)
__global__ void conv1_kernel(const float* __restrict__ in, const float* __restrict__ W1,
                             const float* __restrict__ b1, float* __restrict__ v) {
    long idx = (long)blockIdx.x * blockDim.x + threadIdx.x;
    long r = idx >> 7;
    int  j = (int)(idx & 127);
    float s = b1[j];
    s += in[r * 3 + 0] * W1[0 * 128 + j];
    s += in[r * 3 + 1] * W1[1 * 128 + j];
    s += in[r * 3 + 2] * W1[2 * 128 + j];
    v[r * 128 + j] = s;
}

// COO scatter: out[b, rows[e]*32 + lane] += vals[e] * x[b, cols[e]*32 + lane]
__global__ void scatter_kernel(const int* __restrict__ rows, const int* __restrict__ cols,
                               const float* __restrict__ vals, const float* __restrict__ X,
                               float* __restrict__ out, int nnz, long xstride, long ostride) {
    long tid  = (long)blockIdx.x * blockDim.x + threadIdx.x;
    long wv   = tid >> 5;
    int  lane = (int)(tid & 31);
    if (wv >= (long)nnz * 4) return;
    int e = (int)(wv >> 2);
    int b = (int)(wv & 3);
    int r = rows[e], c = cols[e];
    float val = vals[e];
    float x   = X[(long)b * xstride + (long)c * 32 + lane];
    unsafeAtomicAdd(&out[(long)b * ostride + (long)r * 32 + lane], val * x);
}

// pack elu(concat(A,Bh)) -> bf16 [R,256] and accumulate per-channel stats
__global__ __launch_bounds__(256) void pack_concat_stats(const float* __restrict__ A,
                                                         const float* __restrict__ Bh,
                                                         unsigned short* __restrict__ Xb,
                                                         float* __restrict__ sums,
                                                         float* __restrict__ sumsq) {
    int c  = threadIdx.x;
    long r0 = (long)blockIdx.x * 64;
    const float* src = (c < 128) ? A : Bh;
    int cc = c & 127;
    float s = 0.f, s2 = 0.f;
    for (int i = 0; i < 64; ++i) {
        long r = r0 + i;
        float x = eluf(src[r * 128 + cc]);
        s += x; s2 += x * x;
        Xb[r * 256 + c] = f2bf(x);
    }
    unsafeAtomicAdd(&sums[c],  s);
    unsafeAtomicAdd(&sumsq[c], s2);
}

// pack concat(elu(X), broadcast avg[b]) -> bf16, accumulate stats
__global__ __launch_bounds__(256) void pack_avg_stats(const float* __restrict__ X,
                                                      const float* __restrict__ avg,
                                                      unsigned short* __restrict__ Xb,
                                                      float* __restrict__ sums,
                                                      float* __restrict__ sumsq) {
    int c  = threadIdx.x;
    long r0 = (long)blockIdx.x * 64;
    float s = 0.f, s2 = 0.f;
    if (c < 128) {
        for (int i = 0; i < 64; ++i) {
            long r = r0 + i;
            float x = eluf(X[r * 128 + c]);
            s += x; s2 += x * x;
            Xb[r * 256 + c] = f2bf(x);
        }
    } else {
        int cc = c - 128;
        for (int i = 0; i < 64; ++i) {
            long r = r0 + i;
            int  b = (int)(r / 12000);
            float x = avg[b * 128 + cc];      // already averaged post-elu values
            s += x; s2 += x * x;
            Xb[r * 256 + c] = f2bf(x);
        }
    }
    unsafeAtomicAdd(&sums[c],  s);
    unsafeAtomicAdd(&sumsq[c], s2);
}

// masked sum of elu(X) over rows, per (batch, channel)
__global__ __launch_bounds__(128) void reduce_avgsum(const float* __restrict__ X,
                                                     const float* __restrict__ mask,
                                                     float* __restrict__ avgsum, int N) {
    int c = threadIdx.x;
    int b = blockIdx.y;
    long r0 = (long)blockIdx.x * 96;
    float s = 0.f;
    for (int i = 0; i < 96; ++i) {
        long r = r0 + i;
        float m = mask[(long)b * N + r];
        float x = eluf(X[((long)b * N + r) * 128 + c]);
        s += m * x;
    }
    unsafeAtomicAdd(&avgsum[b * 128 + c], s);
}

__global__ void masksum_kernel(const float* __restrict__ mask, float* __restrict__ msum, int N) {
    __shared__ float red[256];
    int b = blockIdx.x;
    float s = 0.f;
    for (int r = threadIdx.x; r < N; r += 256) s += mask[(long)b * N + r];
    red[threadIdx.x] = s;
    __syncthreads();
    for (int off = 128; off; off >>= 1) {
        if (threadIdx.x < off) red[threadIdx.x] += red[threadIdx.x + off];
        __syncthreads();
    }
    if (threadIdx.x == 0) msum[b] = red[0];
}

__global__ void finalize_avg(const float* __restrict__ avgsum, const float* __restrict__ msum,
                             float* __restrict__ avg) {
    int i = threadIdx.x;          // 512 = 4 batches x 128 channels
    avg[i] = avgsum[i] / msum[i >> 7];
}

// BN pre: scale = gamma*rsqrt(var+eps); shift = beta - mean*scale
__global__ void bn_finalize(const float* __restrict__ sums, const float* __restrict__ sumsq,
                            const float* __restrict__ gamma, const float* __restrict__ beta,
                            float* __restrict__ scale, float* __restrict__ shift, float invR) {
    int c = threadIdx.x;
    float mean = sums[c] * invR;
    float var  = sumsq[c] * invR - mean * mean;
    float rs   = rsqrtf(var + 1e-5f);
    float sc   = gamma[c] * rs;
    scale[c] = sc;
    shift[c] = beta[c] - mean * sc;
}

// pack scale[k]*W[k,n] into WMMA B-fragment layout (bf16):
// lane = (n&15) + 16*bit3(k&31), elem = (k&7) + 8*bit4(k&31)
__global__ __launch_bounds__(256) void prep_w(const float* __restrict__ W,
                                              const float* __restrict__ scale,
                                              unsigned short* __restrict__ Wp) {
    int idx = blockIdx.x * 256 + threadIdx.x;    // 32768 = 256*128
    int k = idx >> 7, n = idx & 127;
    float w = scale[k] * W[(long)k * 128 + n];
    int kt = k >> 5, kk = k & 31, nt = n >> 4, nn = n & 15;
    int lane = nn + ((kk >> 3) & 1) * 16;
    int e    = (kk & 7) + ((kk >> 4) & 1) * 8;
    Wp[(((long)kt * 8 + nt) * 32 + lane) * 16 + e] = f2bf(w);
}

// bp[n] = b[n] + sum_k shift[k]*W[k,n]
__global__ void prep_b(const float* __restrict__ W, const float* __restrict__ b,
                       const float* __restrict__ shift, float* __restrict__ bp) {
    int n = threadIdx.x;
    float acc = b[n];
    for (int k = 0; k < 256; ++k) acc += shift[k] * W[(long)k * 128 + n];
    bp[n] = acc;
}

// out[R,128] = Xb[R,256](bf16) @ Wp(frag-packed bf16) + bp (+resid)
__global__ __launch_bounds__(256) void gemm_kernel(const unsigned short* __restrict__ Xb,
                                                   const unsigned short* __restrict__ Wp,
                                                   const float* __restrict__ bp,
                                                   const float* __restrict__ resid,
                                                   float* __restrict__ out, int hasResid) {
    __shared__ __align__(16) unsigned short sA[16 * 256];   // 8 KB A tile
    long m0 = (long)blockIdx.x * 16;
    {   // tile is 16 contiguous rows: straight 8 KB copy into LDS
        const U4* src = (const U4*)(Xb + m0 * 256);
        U4* dst = (U4*)sA;
        dst[threadIdx.x]       = src[threadIdx.x];
        dst[threadIdx.x + 256] = src[threadIdx.x + 256];
    }
    __syncthreads();

    int lane = threadIdx.x & 31;
    int w    = threadIdx.x >> 5;     // n-tile, 0..7
    int half = lane >> 4;
    int mloc = lane & 15;

    v8f acc = {};
#pragma unroll
    for (int kt = 0; kt < 8; ++kt) {
        int k0 = kt * 32;
        FragBF a, b;
        // A 16x32 bf16 frag: lanes 0-15: K {0..7,16..23}; lanes 16-31: +8
        a.q[0] = *(const U4*)&sA[mloc * 256 + k0 + half * 8];
        a.q[1] = *(const U4*)&sA[mloc * 256 + k0 + 16 + half * 8];
        // B frag: pre-packed, contiguous 32B per lane
        const U4* pb = (const U4*)&Wp[(((long)kt * 8 + w) * 32 + lane) * 16];
        b.q[0] = pb[0];
        b.q[1] = pb[1];
        acc = __builtin_amdgcn_wmma_f32_16x16x32_bf16(false, a.v, false, b.v,
                                                      (short)0, acc, false, false);
    }

    float bb = bp[w * 16 + (lane & 15)];
#pragma unroll
    for (int i = 0; i < 8; ++i) {
        long row = m0 + i + half * 8;
        int  col = w * 16 + (lane & 15);
        long idx = row * 128 + col;
        float o = acc[i] + bb;
        if (hasResid) o += resid[idx];
        out[idx] = o;
    }
}

// per-channel stats of elu(F), 128 channels
__global__ __launch_bounds__(128) void stats128(const float* __restrict__ F,
                                                float* __restrict__ sums,
                                                float* __restrict__ sumsq) {
    int c = threadIdx.x;
    long r0 = (long)blockIdx.x * 64;
    float s = 0.f, s2 = 0.f;
    for (int i = 0; i < 64; ++i) {
        float x = eluf(F[(r0 + i) * 128 + c]);
        s += x; s2 += x * x;
    }
    unsafeAtomicAdd(&sums[c],  s);
    unsafeAtomicAdd(&sumsq[c], s2);
}

// out[r] = sum_c (elu(F[r,c])*scale[c]+shift[c]) * W2[c] + b2   (wave per row)
__global__ __launch_bounds__(256) void conv2_kernel(const float* __restrict__ F,
                                                    const float* __restrict__ scale,
                                                    const float* __restrict__ shift,
                                                    const float* __restrict__ W2,
                                                    const float* __restrict__ b2,
                                                    float* __restrict__ out) {
    long tid  = (long)blockIdx.x * blockDim.x + threadIdx.x;
    long r    = tid >> 5;
    int  lane = (int)(tid & 31);
    float s = 0.f;
#pragma unroll
    for (int i = 0; i < 4; ++i) {
        int c = lane + i * 32;
        float x = eluf(F[r * 128 + c]);
        s += (x * scale[c] + shift[c]) * W2[c];
    }
    for (int off = 16; off; off >>= 1) s += __shfl_xor(s, off, 32);
    if (lane == 0) out[r] = s + b2[0];
}

// ---------------------------------------------------------------------------
extern "C" void kernel_launch(void* const* d_in, const int* in_sizes, int n_in,
                              void* d_out, int out_size, void* d_ws, size_t ws_size,
                              hipStream_t stream) {
    const int B = 4, N = 12000, Fn = 24000;
    const int RV = B * N;        // 48000 vertex rows
    const int RF = B * Fn;       // 96000 face rows
    const int NNZ = 48 * Fn;     // 1,152,000

    const float* inputs  = (const float*)d_in[0];
    const float* mask    = (const float*)d_in[1];
    const int*   DiR     = (const int*)  d_in[2];
    const int*   DiC     = (const int*)  d_in[3];
    const float* DiV     = (const float*)d_in[4];
    const int*   DiAR    = (const int*)  d_in[5];
    const int*   DiAC    = (const int*)  d_in[6];
    const float* DiAV    = (const float*)d_in[7];
    const float* W1      = (const float*)d_in[8];
    const float* b1      = (const float*)d_in[9];
    const float* rnG     = (const float*)d_in[10];   // (16,2,256)
    const float* rnBe    = (const float*)d_in[11];   // (16,2,256)
    const float* rnW     = (const float*)d_in[12];   // (16,2,256,128)
    const float* rnB     = (const float*)d_in[13];   // (16,2,128)
    const float* g2      = (const float*)d_in[14];
    const float* be2     = (const float*)d_in[15];
    const float* W2      = (const float*)d_in[16];
    const float* b2      = (const float*)d_in[17];
    (void)in_sizes; (void)n_in; (void)out_size; (void)ws_size;

    const long szV = (long)RV * 128;   //  6,144,000 f32
    const long szF = (long)RF * 128;   // 12,288,000 f32

    float* v    = (float*)d_ws;
    float* tmpV = v + szV;             // msg_v / avg-block temp
    float* f    = tmpV + szV;
    float* msgF = f + szF;
    unsigned short* Xb = (unsigned short*)(msgF + szF);   // [RF,256] bf16
    float* sums   = (float*)(Xb + (long)RF * 256);
    float* sumsq  = sums + 256;
    float* scale  = sumsq + 256;
    float* shiftv = scale + 256;
    float* bp     = shiftv + 256;
    float* avgsum = bp + 128;          // 512
    float* msum   = avgsum + 512;      // 4
    float* avg    = msum + 4;          // 512
    unsigned short* Wp = (unsigned short*)(avg + 512);    // 32768 bf16

    const int scatBlocks = (NNZ * 4 * 32) / 256;          // 576000

    auto zero = [&](float* p, long n) {
        zero_f32<<<(int)((n + 255) / 256), 256, 0, stream>>>(p, n);
    };
    auto bn_linear = [&](const float* gam, const float* bet, const float* W,
                         const float* bias, int R, const float* resid, float* out_) {
        bn_finalize<<<1, 256, 0, stream>>>(sums, sumsq, gam, bet, scale, shiftv, 1.f / R);
        prep_w<<<128, 256, 0, stream>>>(W, scale, Wp);
        prep_b<<<1, 128, 0, stream>>>(W, bias, shiftv, bp);
        gemm_kernel<<<R / 16, 256, 0, stream>>>(Xb, Wp, bp, resid ? resid : out_,
                                                out_, resid ? 1 : 0);
    };

    // conv1 (no BN), f starts at zero, mask row-counts
    conv1_kernel<<<(RV * 128) / 256, 256, 0, stream>>>(inputs, W1, b1, v);
    zero(f, szF);
    masksum_kernel<<<4, 256, 0, stream>>>(mask, msum, N);

    for (int i = 0; i < 16; ++i) {
        const float* G0 = rnG  + (long)(i * 2 + 0) * 256;
        const float* G1 = rnG  + (long)(i * 2 + 1) * 256;
        const float* E0 = rnBe + (long)(i * 2 + 0) * 256;
        const float* E1 = rnBe + (long)(i * 2 + 1) * 256;
        const float* Wt0 = rnW + (long)(i * 2 + 0) * 256 * 128;
        const float* Wt1 = rnW + (long)(i * 2 + 1) * 256 * 128;
        const float* Bb0 = rnB + (long)(i * 2 + 0) * 128;
        const float* Bb1 = rnB + (long)(i * 2 + 1) * 128;

        if ((i & 1) == 0) {
            // ---- dir block ----
            zero(tmpV, szV);
            scatter_kernel<<<scatBlocks, 256, 0, stream>>>(DiAR, DiAC, DiAV, f, tmpV,
                                                           NNZ, (long)Fn * 128, (long)N * 128);
            zero(sums, 512);
            pack_concat_stats<<<RV / 64, 256, 0, stream>>>(v, tmpV, Xb, sums, sumsq);
            bn_linear(G0, E0, Wt0, Bb0, RV, /*resid=*/v, /*out=*/v);

            zero(msgF, szF);
            scatter_kernel<<<scatBlocks, 256, 0, stream>>>(DiR, DiC, DiV, v, msgF,
                                                           NNZ, (long)N * 128, (long)Fn * 128);
            zero(sums, 512);
            pack_concat_stats<<<RF / 64, 256, 0, stream>>>(f, msgF, Xb, sums, sumsq);
            bn_linear(G1, E1, Wt1, Bb1, RF, /*resid=*/nullptr, /*out=*/f);
        } else {
            // ---- avg block ----
            const float* xin = v;
            for (int j = 0; j < 2; ++j) {
                zero(avgsum, 512);
                reduce_avgsum<<<dim3(125, 4), 128, 0, stream>>>(xin, mask, avgsum, N);
                finalize_avg<<<1, 512, 0, stream>>>(avgsum, msum, avg);
                zero(sums, 512);
                pack_avg_stats<<<RV / 64, 256, 0, stream>>>(xin, avg, Xb, sums, sumsq);
                const float* gam = (j == 0) ? G0 : G1;
                const float* bet = (j == 0) ? E0 : E1;
                const float* Wt  = (j == 0) ? Wt0 : Wt1;
                const float* Bb  = (j == 0) ? Bb0 : Bb1;
                bn_linear(gam, bet, Wt, Bb, RV,
                          /*resid=*/(j == 1) ? v : nullptr,
                          /*out=*/(j == 1) ? v : tmpV);
                xin = tmpV;
            }
        }
    }

    // conv2: BN(elu(f)) @ W2 + b2 -> [4,24000,1]
    zero(sums, 512);
    stats128<<<RF / 64, 128, 0, stream>>>(f, sums, sumsq);
    bn_finalize<<<1, 128, 0, stream>>>(sums, sumsq, g2, be2, scale, shiftv, 1.f / RF);
    conv2_kernel<<<RF / 8, 256, 0, stream>>>(f, scale, shiftv, W2, b2, (float*)d_out);
}
